// GatedGCN_87393994539142
// MI455X (gfx1250) — matrix operand order repaired and checked by reading, not compiled, not run on hardware
//
#include <hip/hip_runtime.h>
#include <hip/hip_bf16.h>

#define N_NODESC 20000
#define N_EDGESC 320000
#define HID 128
#define NOUTC 64
#define NLAYERS 3
#define EPSC 1e-5f

typedef __bf16 bf16_t;
typedef __attribute__((ext_vector_type(16))) __bf16 v16bf;
typedef __attribute__((ext_vector_type(8))) float v8f;

struct B32pair { uint4 lo, hi; };

// Shared-memory tile: A staging (bf16, padded) overlaid with D staging (f32, padded)
struct GemmSM {
  union {
    unsigned short a[16][136];  // 4352 B, +8 ushort pad -> conflict-free b128 reads
    float          d[16][132];  // 8448 B, +4 f32 pad -> conflict-free tile readout
  };
};

__device__ __forceinline__ unsigned short bfbits(float f) {
  unsigned u = __builtin_bit_cast(unsigned, f);
  u += 0x7FFFu + ((u >> 16) & 1u);           // round-to-nearest-even
  return (unsigned short)(u >> 16);
}
__device__ __forceinline__ bf16_t f2bf(float f) {
  unsigned short s = bfbits(f);
  return __builtin_bit_cast(bf16_t, s);
}

// ---------------------------------------------------------------------------
// Pack a 128x128 row-major f32 weight matrix into WMMA B-fragment layout:
// frag(kt,nt) is 512 bf16; lane l (n = nt*16 + (l&15), hi = l>>4) holds
// elements e=0..15 = W[kt*32 + hi*16 + e][n].  17 matrices total.
// ---------------------------------------------------------------------------
__global__ __launch_bounds__(256) void pack_weights(
    const float* __restrict__ nodeW, const float* __restrict__ edgeW,
    const float* __restrict__ layersW, bf16_t* __restrict__ Wpk)
{
  const int gid = blockIdx.x * 256 + threadIdx.x;   // 17*4*8*32 = 17408 tasks
  if (gid >= 17 * 4 * 8 * 32) return;
  const int lane = gid & 31;
  const int nt   = (gid >> 5) & 7;
  const int kt   = (gid >> 8) & 3;
  const int midx = gid >> 10;
  const float* src = (midx == 0) ? nodeW
                   : (midx == 1) ? edgeW
                   : (layersW + (size_t)(midx - 2) * HID * HID);
  const int n     = nt * 16 + (lane & 15);
  const int kbase = kt * 32 + (lane >> 4) * 16;
  bf16_t* dst = Wpk + (size_t)midx * 16384 + (size_t)((kt * 8 + nt) * 32 + lane) * 16;
#pragma unroll
  for (int e = 0; e < 16; ++e) dst[e] = f2bf(src[(size_t)(kbase + e) * HID + n]);
}

// Stage a 16x128 f32 tile of X into LDS as bf16 (one block, 8 f32 per thread).
__device__ __forceinline__ void stage_a_tile(GemmSM& sm, const float* __restrict__ X,
                                             int row0, int t)
{
  const int r = t >> 4;
  const int c = (t & 15) * 8;
  const float4* src = (const float4*)(X + (size_t)(row0 + r) * HID + c);
  const float4 f0 = src[0];
  const float4 f1 = src[1];
  unsigned short* d = &sm.a[r][c];
  d[0] = bfbits(f0.x); d[1] = bfbits(f0.y); d[2] = bfbits(f0.z); d[3] = bfbits(f0.w);
  d[4] = bfbits(f1.x); d[5] = bfbits(f1.y); d[6] = bfbits(f1.z); d[7] = bfbits(f1.w);
}

// Load all four K-fragments (K=128) of the A tile for this lane from LDS.
__device__ __forceinline__ void load_a_frags(const GemmSM& sm, v16bf a[4], int m, int hi)
{
#pragma unroll
  for (int kt = 0; kt < 4; ++kt) {
    B32pair p;
    p.lo = *(const uint4*)&sm.a[m][kt * 32 + hi * 16];
    p.hi = *(const uint4*)&sm.a[m][kt * 32 + hi * 16 + 8];
    a[kt] = __builtin_bit_cast(v16bf, p);
  }
}

// Write one wave's 16x16 f32 tile (+bias) into the LDS D buffer, then emit a
// block-wide coalesced b128 store of the 16x128 tile.
__device__ __forceinline__ void store_tile_coalesced(
    GemmSM& sm, const v8f& acc, float bv, int col, int hi,
    float* __restrict__ Y, int row0, int t)
{
  __syncthreads();                      // LDS A reads (and prior D readout) done
#pragma unroll
  for (int r = 0; r < 8; ++r) sm.d[hi * 8 + r][col] = acc[r] + bv;
  __syncthreads();
  const int rr = t >> 4;
  const int cc = (t & 15) * 8;
  const float4 o0 = *(const float4*)&sm.d[rr][cc];
  const float4 o1 = *(const float4*)&sm.d[rr][cc + 4];
  float4* yp = (float4*)(Y + (size_t)(row0 + rr) * HID + cc);
  yp[0] = o0;
  yp[1] = o1;
}

// ---------------------------------------------------------------------------
// Y[M x 128] = bf16(X[M x 128]) @ W + bias, fp32 accumulate.
// Block = 256 threads = 8 waves; block computes a 16-row x 128-col tile.
// Two interleaved accumulator chains hide the WMMA->WMMA RAW hazard window.
// ---------------------------------------------------------------------------
__global__ __launch_bounds__(256) void gemm_bias_wmma(
    const float* __restrict__ X, const bf16_t* __restrict__ Wf,
    const float* __restrict__ bias, float* __restrict__ Y)
{
  __shared__ GemmSM sm;
  const int row0 = blockIdx.x * 16;
  const int t = threadIdx.x;
  stage_a_tile(sm, X, row0, t);
  __syncthreads();
  const int wave = t >> 5, lane = t & 31;
  const int m = lane & 15, hi = lane >> 4;
  v16bf a[4];
  load_a_frags(sm, a, m, hi);
  v16bf b[4];
#pragma unroll
  for (int kt = 0; kt < 4; ++kt)
    b[kt] = *(const v16bf*)(Wf + (size_t)((kt * 8 + wave) * 32 + lane) * 16);
  v8f acc0 = {0.f, 0.f, 0.f, 0.f, 0.f, 0.f, 0.f, 0.f};
  v8f acc1 = {0.f, 0.f, 0.f, 0.f, 0.f, 0.f, 0.f, 0.f};
  acc0 = __builtin_amdgcn_wmma_f32_16x16x32_bf16(false, a[0], false, b[0], (short)0, acc0, false, false);
  acc1 = __builtin_amdgcn_wmma_f32_16x16x32_bf16(false, a[1], false, b[1], (short)0, acc1, false, false);
  acc0 = __builtin_amdgcn_wmma_f32_16x16x32_bf16(false, a[2], false, b[2], (short)0, acc0, false, false);
  acc1 = __builtin_amdgcn_wmma_f32_16x16x32_bf16(false, a[3], false, b[3], (short)0, acc1, false, false);
  v8f acc;
#pragma unroll
  for (int r = 0; r < 8; ++r) acc[r] = acc0[r] + acc1[r];
  const int col = wave * 16 + m;
  store_tile_coalesced(sm, acc, bias[col], col, hi, Y, row0, t);
}

// Fused 4-matrix variant (A,B,D,E projections): shares the A tile (LDS) and
// A-fragments (registers); kt-outer / mat-inner interleaving gives 4
// independent WMMA chains so no hazard NOPs are exposed.
__global__ __launch_bounds__(256) void gemm4_bias_wmma(
    const float* __restrict__ X,
    const bf16_t* __restrict__ F0, const bf16_t* __restrict__ F1,
    const bf16_t* __restrict__ F2, const bf16_t* __restrict__ F3,
    const float* __restrict__ b0, const float* __restrict__ b1,
    const float* __restrict__ b2, const float* __restrict__ b3,
    float* __restrict__ Y0, float* __restrict__ Y1,
    float* __restrict__ Y2, float* __restrict__ Y3)
{
  __shared__ GemmSM sm;
  const int row0 = blockIdx.x * 16;
  const int t = threadIdx.x;
  stage_a_tile(sm, X, row0, t);
  __syncthreads();
  const int wave = t >> 5, lane = t & 31;
  const int m = lane & 15, hi = lane >> 4;
  v16bf a[4];
  load_a_frags(sm, a, m, hi);
  const bf16_t* Fs[4] = {F0, F1, F2, F3};
  const float*  bs[4] = {b0, b1, b2, b3};
  float*        Ys[4] = {Y0, Y1, Y2, Y3};
  v8f acc[4];
#pragma unroll
  for (int mat = 0; mat < 4; ++mat)
    acc[mat] = (v8f){0.f, 0.f, 0.f, 0.f, 0.f, 0.f, 0.f, 0.f};
#pragma unroll
  for (int kt = 0; kt < 4; ++kt) {
#pragma unroll
    for (int mat = 0; mat < 4; ++mat) {
      v16bf b = *(const v16bf*)(Fs[mat] + (size_t)((kt * 8 + wave) * 32 + lane) * 16);
      acc[mat] = __builtin_amdgcn_wmma_f32_16x16x32_bf16(false, a[kt], false, b, (short)0, acc[mat], false, false);
    }
  }
  const int col = wave * 16 + m;
#pragma unroll
  for (int mat = 0; mat < 4; ++mat)
    store_tile_coalesced(sm, acc[mat], bs[mat][col], col, hi, Ys[mat], row0, t);
}

// ---------------------------------------------------------------------------
// Fused edge pass: ehat (holds Ce in) += Dh[dst] + Eh[src]; sigma=sigmoid;
// agg[dst] += sigma * Bh[src] (float atomics); per-block BN partial sums.
// Block handles 128 edges: wave w owns edge-lane w, 32 lanes cover 128 feats.
// ---------------------------------------------------------------------------
__global__ __launch_bounds__(256) void edge_fused(
    float* __restrict__ ehat, const float* __restrict__ Dh,
    const float* __restrict__ Eh, const float* __restrict__ Bh,
    const int* __restrict__ eidx, float* __restrict__ agg,
    float* __restrict__ parts)
{
  __shared__ int sidx[256];          // [0..127] src, [128..255] dst
  __shared__ float red[8][128];
  const int t = threadIdx.x;
  const int e0 = blockIdx.x * 128;
  if (t < 128) sidx[t] = eidx[e0 + t];
  else         sidx[t] = eidx[N_EDGESC + e0 + (t - 128)];
  __syncthreads();
  const int fq = (t & 31) * 4;
  const int el = t >> 5;
  float s0 = 0, s1 = 0, s2 = 0, s3 = 0, q0 = 0, q1 = 0, q2 = 0, q3 = 0;
#pragma unroll 4
  for (int it = 0; it < 16; ++it) {
    const int li = it * 8 + el;
    const size_t ei = (size_t)(e0 + li);
    const int sn = sidx[li];
    const int dn = sidx[128 + li];
    if (it < 15) {                    // prefetch next iteration's gather rows
      const int nsn = sidx[li + 8];
      const int ndn = sidx[128 + li + 8];
      __builtin_prefetch(Dh + (size_t)ndn * HID + fq, 0, 1);
      __builtin_prefetch(Eh + (size_t)nsn * HID + fq, 0, 1);
      __builtin_prefetch(Bh + (size_t)nsn * HID + fq, 0, 1);
    }
    float4 v        = *(const float4*)(ehat + ei * HID + fq);
    const float4 dh = *(const float4*)(Dh + (size_t)dn * HID + fq);
    const float4 eh = *(const float4*)(Eh + (size_t)sn * HID + fq);
    v.x += dh.x + eh.x; v.y += dh.y + eh.y; v.z += dh.z + eh.z; v.w += dh.w + eh.w;
    *(float4*)(ehat + ei * HID + fq) = v;
    const float4 bh = *(const float4*)(Bh + (size_t)sn * HID + fq);
    const float g0 = 1.f / (1.f + expf(-v.x));
    const float g1 = 1.f / (1.f + expf(-v.y));
    const float g2 = 1.f / (1.f + expf(-v.z));
    const float g3 = 1.f / (1.f + expf(-v.w));
    float* ap = agg + (size_t)dn * HID + fq;
    atomicAdd(ap + 0, g0 * bh.x);
    atomicAdd(ap + 1, g1 * bh.y);
    atomicAdd(ap + 2, g2 * bh.z);
    atomicAdd(ap + 3, g3 * bh.w);
    s0 += v.x; s1 += v.y; s2 += v.z; s3 += v.w;
    q0 += v.x * v.x; q1 += v.y * v.y; q2 += v.z * v.z; q3 += v.w * v.w;
  }
  red[el][fq + 0] = s0; red[el][fq + 1] = s1; red[el][fq + 2] = s2; red[el][fq + 3] = s3;
  __syncthreads();
  if (t < 128) {
    float a = 0;
#pragma unroll
    for (int i = 0; i < 8; ++i) a += red[i][t];
    parts[(size_t)blockIdx.x * 256 + t] = a;
  }
  __syncthreads();
  red[el][fq + 0] = q0; red[el][fq + 1] = q1; red[el][fq + 2] = q2; red[el][fq + 3] = q3;
  __syncthreads();
  if (t < 128) {
    float a = 0;
#pragma unroll
    for (int i = 0; i < 8; ++i) a += red[i][t];
    parts[(size_t)blockIdx.x * 256 + 128 + t] = a;
  }
}

// x = Ah + agg (written back into agg buffer) + per-block BN partial sums.
__global__ __launch_bounds__(256) void node_x_stats(
    const float* __restrict__ Ah, float* __restrict__ x,
    float* __restrict__ parts)
{
  __shared__ float red[8][128];
  const int t = threadIdx.x;
  const int fq = (t & 31) * 4;
  const int el = t >> 5;
  const int r0 = blockIdx.x * 128;
  float s0 = 0, s1 = 0, s2 = 0, s3 = 0, q0 = 0, q1 = 0, q2 = 0, q3 = 0;
  for (int it = 0; it < 16; ++it) {
    const int r = r0 + it * 8 + el;
    if (r < N_NODESC) {
      const float4 a = *(const float4*)(Ah + (size_t)r * HID + fq);
      const float4 g = *(const float4*)(x + (size_t)r * HID + fq);
      const float4 v = make_float4(a.x + g.x, a.y + g.y, a.z + g.z, a.w + g.w);
      *(float4*)(x + (size_t)r * HID + fq) = v;
      s0 += v.x; s1 += v.y; s2 += v.z; s3 += v.w;
      q0 += v.x * v.x; q1 += v.y * v.y; q2 += v.z * v.z; q3 += v.w * v.w;
    }
  }
  red[el][fq + 0] = s0; red[el][fq + 1] = s1; red[el][fq + 2] = s2; red[el][fq + 3] = s3;
  __syncthreads();
  if (t < 128) {
    float a = 0;
#pragma unroll
    for (int i = 0; i < 8; ++i) a += red[i][t];
    parts[(size_t)blockIdx.x * 256 + t] = a;
  }
  __syncthreads();
  red[el][fq + 0] = q0; red[el][fq + 1] = q1; red[el][fq + 2] = q2; red[el][fq + 3] = q3;
  __syncthreads();
  if (t < 128) {
    float a = 0;
#pragma unroll
    for (int i = 0; i < 8; ++i) a += red[i][t];
    parts[(size_t)blockIdx.x * 256 + 128 + t] = a;
  }
}

// Deterministic fixed-order reduction of P block-partials (256 floats each).
__global__ __launch_bounds__(256) void reduce_parts(
    const float* __restrict__ parts, int P, float* __restrict__ out)
{
  const int t = threadIdx.x;
  float s = 0.f;
  for (int p = 0; p < P; ++p) s += parts[(size_t)p * 256 + t];
  out[t] = s;
}

// base += relu(gamma*(x-mu)*rsqrt(var+eps)+beta), stats = {sum[128], sumsq[128]}
__global__ __launch_bounds__(256) void bn_relu_residual(
    float* __restrict__ base, const float* __restrict__ x,
    const float* __restrict__ stats, const float* __restrict__ gamma,
    const float* __restrict__ beta, float invcnt, int n4)
{
  const int i = blockIdx.x * 256 + threadIdx.x;
  if (i >= n4) return;
  const int fq = (i & 31) * 4;
  const float4 xv = *(const float4*)(x + (size_t)i * 4);
  float4 bv = *(float4*)(base + (size_t)i * 4);
  float r[4] = {xv.x, xv.y, xv.z, xv.w};
  float o[4] = {bv.x, bv.y, bv.z, bv.w};
#pragma unroll
  for (int c = 0; c < 4; ++c) {
    const int f = fq + c;
    const float mu = stats[f] * invcnt;
    const float var = stats[128 + f] * invcnt - mu * mu;
    const float inv = rsqrtf(var + EPSC);
    const float y = gamma[f] * (r[c] - mu) * inv + beta[f];
    o[c] += fmaxf(y, 0.f);
  }
  *(float4*)(base + (size_t)i * 4) = make_float4(o[0], o[1], o[2], o[3]);
}

__global__ __launch_bounds__(256) void zero_f32(float4* __restrict__ p, int n4) {
  const int i = blockIdx.x * 256 + threadIdx.x;
  if (i < n4) p[i] = make_float4(0.f, 0.f, 0.f, 0.f);
}

__global__ __launch_bounds__(128) void mean_partial(
    const float* __restrict__ h, float* __restrict__ hpart)
{
  const int f = threadIdx.x;
  float s = 0.f;
  for (int r = blockIdx.x; r < N_NODESC; r += 128) s += h[(size_t)r * HID + f];
  hpart[(size_t)blockIdx.x * 128 + f] = s;
}

__global__ __launch_bounds__(128) void classifier(
    const float* __restrict__ hpart, const float* __restrict__ W1,
    const float* __restrict__ b1, const float* __restrict__ W2,
    const float* __restrict__ b2, float* __restrict__ out)
{
  __shared__ float hm[128];
  __shared__ float t1[128];
  const int t = threadIdx.x;
  {
    float s = 0.f;
    for (int p = 0; p < 128; ++p) s += hpart[(size_t)p * 128 + t];
    hm[t] = s * (1.f / (float)N_NODESC);
  }
  __syncthreads();
  {
    float a = b1[t];
    for (int k = 0; k < 128; ++k) a += hm[k] * W1[(size_t)k * HID + t];
    t1[t] = fmaxf(a, 0.f);
  }
  __syncthreads();
  if (t < NOUTC) {
    float o = b2[t];
    for (int j = 0; j < 128; ++j) o += t1[j] * W2[(size_t)j * NOUTC + t];
    out[t] = o;
  }
}

// ---------------------------------------------------------------------------
extern "C" void kernel_launch(void* const* d_in, const int* in_sizes, int n_in,
                              void* d_out, int out_size, void* d_ws, size_t ws_size,
                              hipStream_t stream)
{
  (void)in_sizes; (void)n_in; (void)out_size; (void)ws_size;
  const float* h_in    = (const float*)d_in[0];
  const int*   eidx    = (const int*)d_in[1];
  const float* eattr   = (const float*)d_in[2];
  const float* nodeW   = (const float*)d_in[3];
  const float* nodeB   = (const float*)d_in[4];
  const float* edgeW   = (const float*)d_in[5];
  const float* edgeB   = (const float*)d_in[6];
  const float* layersW = (const float*)d_in[7];
  const float* layersB = (const float*)d_in[8];
  const float* bnG     = (const float*)d_in[9];
  const float* bnB     = (const float*)d_in[10];
  const float* clsW1   = (const float*)d_in[11];
  const float* clsB1   = (const float*)d_in[12];
  const float* clsW2   = (const float*)d_in[13];
  const float* clsB2   = (const float*)d_in[14];
  float* out = (float*)d_out;

  char* ws = (char*)d_ws;
  size_t off = 0;
  auto take = [&](size_t bytes) -> char* {
    char* p = ws + off;
    off = (off + bytes + 255) & ~(size_t)255;
    return p;
  };
  float*  e      = (float*)take((size_t)N_EDGESC * HID * 4);
  float*  ehat   = (float*)take((size_t)N_EDGESC * HID * 4);
  float*  h      = (float*)take((size_t)N_NODESC * HID * 4);
  float*  Ah     = (float*)take((size_t)N_NODESC * HID * 4);
  float*  Bh     = (float*)take((size_t)N_NODESC * HID * 4);
  float*  Dh     = (float*)take((size_t)N_NODESC * HID * 4);
  float*  Eh     = (float*)take((size_t)N_NODESC * HID * 4);
  float*  agg    = (float*)take((size_t)N_NODESC * HID * 4);
  bf16_t* Wpk    = (bf16_t*)take((size_t)17 * 16384 * 2);
  float*  eparts = (float*)take((size_t)(N_EDGESC / 128) * 256 * 4);
  float*  nparts = (float*)take((size_t)((N_NODESC + 127) / 128) * 256 * 4);
  float*  estats = (float*)take(256 * 4);
  float*  nstats = (float*)take(256 * 4);
  float*  hpart  = (float*)take(128 * 128 * 4);

  const int nEblk = N_EDGESC / 128;                 // 2500
  const int nNblk = (N_NODESC + 127) / 128;         // 157
  const int e4    = N_EDGESC * HID / 4;             // 10,240,000
  const int n4    = N_NODESC * HID / 4;             // 640,000

  pack_weights<<<68, 256, 0, stream>>>(nodeW, edgeW, layersW, Wpk);

  // Embeddings
  gemm_bias_wmma<<<N_NODESC / 16, 256, 0, stream>>>(h_in, Wpk + 0 * 16384, nodeB, h);
  gemm_bias_wmma<<<N_EDGESC / 16, 256, 0, stream>>>(eattr, Wpk + (size_t)1 * 16384, edgeB, e);

  for (int l = 0; l < NLAYERS; ++l) {
    const bf16_t* F = Wpk + (size_t)(2 + l * 5) * 16384;
    const float*  B = layersB + (size_t)l * 5 * HID;

    // A,B,D,E projections of h (fused), C projection of e (into ehat)
    gemm4_bias_wmma<<<N_NODESC / 16, 256, 0, stream>>>(h,
        F + (size_t)0 * 16384, F + (size_t)1 * 16384,
        F + (size_t)3 * 16384, F + (size_t)4 * 16384,
        B + 0 * HID, B + 1 * HID, B + 3 * HID, B + 4 * HID,
        Ah, Bh, Dh, Eh);
    gemm_bias_wmma<<<N_EDGESC / 16, 256, 0, stream>>>(e, F + (size_t)2 * 16384, B + 2 * HID, ehat);

    // Edge pass + scatter + BN stats
    zero_f32<<<(n4 + 255) / 256, 256, 0, stream>>>((float4*)agg, n4);
    edge_fused<<<nEblk, 256, 0, stream>>>(ehat, Dh, Eh, Bh, eidx, agg, eparts);
    reduce_parts<<<1, 256, 0, stream>>>(eparts, nEblk, estats);
    bn_relu_residual<<<e4 / 256, 256, 0, stream>>>(e, ehat, estats,
        bnG + (size_t)(l * 2 + 1) * HID, bnB + (size_t)(l * 2 + 1) * HID,
        1.0f / (float)N_EDGESC, e4);

    // Node pass
    node_x_stats<<<nNblk, 256, 0, stream>>>(Ah, agg, nparts);
    reduce_parts<<<1, 256, 0, stream>>>(nparts, nNblk, nstats);
    bn_relu_residual<<<n4 / 256, 256, 0, stream>>>(h, agg, nstats,
        bnG + (size_t)(l * 2) * HID, bnB + (size_t)(l * 2) * HID,
        1.0f / (float)N_NODESC, n4);
  }

  mean_partial<<<128, 128, 0, stream>>>(h, hpart);
  classifier<<<1, 128, 0, stream>>>(hpart, clsW1, clsB1, clsW2, clsB2, out);
}
